// RandomProjection_25778393711209
// MI455X (gfx1250) — matrix operand order repaired and checked by reading, not compiled, not run on hardware
//
#include <hip/hip_runtime.h>
#include <hip/hip_bf16.h>

// Problem: B=32, S=512, D=768, O=2048
//   out[b,o] = mean_s( normalize(x[b,s,:]) . normalize(p[o,:]) )
//            = m[b,:] . pn[o,:]   with  m = (1/S) * sum_s normalize(x[b,s,:])
// => one bandwidth-bound pass over x (50 MB, ~2.4us at 23.3 TB/s), then a tiny
//    WMMA GEMM ([32x768]x[768x2048], ~100 MFLOP) on v_wmma_f32_16x16x32_bf16.

typedef __bf16 bf16;
typedef __attribute__((ext_vector_type(16))) bf16  v16bf;
typedef __attribute__((ext_vector_type(8)))  bf16  v8bf;
typedef __attribute__((ext_vector_type(4)))  bf16  v4bf;
typedef __attribute__((ext_vector_type(8)))  float v8f;
typedef __attribute__((ext_vector_type(4)))  float v4f;

#define B_  32
#define S_  512
#define D_  768
#define O_  2048
#define KV  6    // 768 floats / (32 lanes * 4 per float4)

// ---------------------------------------------------------------------------
// Stage 1: per-row L2 normalize x, accumulate sum over s into m_sum[b][d].
// One wave handles 8 rows; block = 8 waves; grid = B * (S/64) = 256 blocks.
// b128 loads: lane handles d = lane*4 + 128*k -> 512B contiguous per wave-op.
// ---------------------------------------------------------------------------
__global__ __launch_bounds__(256) void rp_reduce_x(const float* __restrict__ x,
                                                   float* __restrict__ m_sum) {
    const int lane  = threadIdx.x & 31;
    const int wave  = threadIdx.x >> 5;
    const int b     = blockIdx.x >> 3;   // 0..31
    const int chunk = blockIdx.x & 7;    // 0..7
    const int s0    = chunk * 64 + wave * 8;

    float acc[KV][4];
#pragma unroll
    for (int k = 0; k < KV; ++k)
#pragma unroll
        for (int j = 0; j < 4; ++j) acc[k][j] = 0.0f;

    for (int r = 0; r < 8; ++r) {
        const v4f* row = (const v4f*)(x + ((size_t)(b * S_ + s0 + r)) * D_) + lane;
        v4f v[KV];
        float ss = 0.0f;
#pragma unroll
        for (int k = 0; k < KV; ++k) {
            v[k] = row[k * 32];
#pragma unroll
            for (int j = 0; j < 4; ++j) ss = fmaf(v[k][j], v[k][j], ss);
        }
        // wave32 butterfly reduction
#pragma unroll
        for (int off = 16; off > 0; off >>= 1) ss += __shfl_xor(ss, off, 32);
        const float inv = 1.0f / fmaxf(sqrtf(ss), 1e-8f);
#pragma unroll
        for (int k = 0; k < KV; ++k)
#pragma unroll
            for (int j = 0; j < 4; ++j) acc[k][j] = fmaf(v[k][j], inv, acc[k][j]);
    }

    float* dst = m_sum + (size_t)b * D_ + lane * 4;
#pragma unroll
    for (int k = 0; k < KV; ++k)
#pragma unroll
        for (int j = 0; j < 4; ++j) unsafeAtomicAdd(dst + k * 128 + j, acc[k][j]);
}

// ---------------------------------------------------------------------------
// Stage 2: normalize projection rows, emit split-bf16 (hi + lo ~= f32 value).
// One wave per row; b128 loads, packed 8-byte bf16x4 stores.
// ---------------------------------------------------------------------------
__global__ __launch_bounds__(256) void rp_norm_p(const float* __restrict__ p,
                                                 bf16* __restrict__ ph,
                                                 bf16* __restrict__ pl) {
    const int lane = threadIdx.x & 31;
    const int wave = threadIdx.x >> 5;
    const int row  = blockIdx.x * 8 + wave;     // 0..2047
    const v4f* src = (const v4f*)(p + (size_t)row * D_) + lane;

    v4f v[KV];
    float ss = 0.0f;
#pragma unroll
    for (int k = 0; k < KV; ++k) {
        v[k] = src[k * 32];
#pragma unroll
        for (int j = 0; j < 4; ++j) ss = fmaf(v[k][j], v[k][j], ss);
    }
#pragma unroll
    for (int off = 16; off > 0; off >>= 1) ss += __shfl_xor(ss, off, 32);
    const float inv = 1.0f / fmaxf(sqrtf(ss), 1e-8f);

    v4bf* dh = (v4bf*)(ph + (size_t)row * D_) + lane;
    v4bf* dl = (v4bf*)(pl + (size_t)row * D_) + lane;
#pragma unroll
    for (int k = 0; k < KV; ++k) {
        v4bf hv, lv;
#pragma unroll
        for (int j = 0; j < 4; ++j) {
            const float val = v[k][j] * inv;
            const bf16  h   = (bf16)val;
            hv[j] = h;
            lv[j] = (bf16)(val - (float)h);
        }
        dh[k * 32] = hv;
        dl[k * 32] = lv;
    }
}

// ---------------------------------------------------------------------------
// Stage 3: m = m_sum / S, split into bf16 hi/lo.  24576 elements.
// ---------------------------------------------------------------------------
__global__ __launch_bounds__(256) void rp_finalize_m(const float* __restrict__ m_sum,
                                                     bf16* __restrict__ mh,
                                                     bf16* __restrict__ ml) {
    const int i = blockIdx.x * 256 + threadIdx.x;   // grid covers exactly B_*D_
    const float val = m_sum[i] * (1.0f / (float)S_);
    const bf16  h   = (bf16)val;
    mh[i] = h;
    ml[i] = (bf16)(val - (float)h);
}

// ---------------------------------------------------------------------------
// Stage 4: out[32,2048] = m[32,768] @ pn[2048,768]^T via v_wmma_f32_16x16x32_bf16.
// Split-precision: A_hi*B_hi + A_hi*B_lo + A_lo*B_hi (lo*lo dropped).
// Fragment layouts used here (wave32):
//   A (16x32, bf16): lane L -> row M=L%16; elems e: K = 16*(L/16) + e
//     => two contiguous 8-elem (16B) loads at K = 16*(L/16) and K = 16*(L/16) + 8
//   B (32x16, bf16): lane L -> col N=L%16; elems e: K = 16*(L/16) + e
//     => one contiguous 16-elem (32B) load; row-major pn[n][k] is exactly B's layout
//   C/D (16x16, f32): VGPR r -> M = r + 8*(L/16), N = L%16
// ---------------------------------------------------------------------------
__device__ __forceinline__ v16bf rp_load_a(const bf16* base) {
    // base already includes row*D + 16*(lane>>4); halves at +0 and +8 K-offset
    const v8bf lo = *(const v8bf*)(base);
    const v8bf hi = *(const v8bf*)(base + 8);
    v16bf r;
#pragma unroll
    for (int i = 0; i < 8; ++i) { r[i] = lo[i]; r[i + 8] = hi[i]; }
    return r;
}

__device__ __forceinline__ v8f rp_wmma(v16bf a, v16bf b, v8f c) {
    return __builtin_amdgcn_wmma_f32_16x16x32_bf16(
        /*neg_a=*/false, a, /*neg_b=*/false, b,
        /*c_mod=*/(short)0, c, /*reuse_a=*/false, /*reuse_b=*/false);
}

__global__ __launch_bounds__(128) void rp_gemm(const bf16* __restrict__ mh,
                                               const bf16* __restrict__ ml,
                                               const bf16* __restrict__ ph,
                                               const bf16* __restrict__ pl,
                                               float* __restrict__ out) {
    const int lane  = threadIdx.x & 31;
    const int wave  = threadIdx.x >> 5;
    const int ntile = blockIdx.x * 4 + wave;    // 0..127
    const int nbase = ntile * 16;
    const int col   = lane & 15;
    const int half  = lane >> 4;

    v8f acc0 = {};
    v8f acc1 = {};

    const bf16* bhp  = ph + (size_t)(nbase + col) * D_ + half * 16;
    const bf16* blp  = pl + (size_t)(nbase + col) * D_ + half * 16;
    const bf16* a0hp = mh + (size_t)col        * D_ + half * 16;
    const bf16* a0lp = ml + (size_t)col        * D_ + half * 16;
    const bf16* a1hp = mh + (size_t)(16 + col) * D_ + half * 16;
    const bf16* a1lp = ml + (size_t)(16 + col) * D_ + half * 16;

    for (int kb = 0; kb < D_; kb += 32) {
        const v16bf bh  = *(const v16bf*)(bhp + kb);
        const v16bf bl  = *(const v16bf*)(blp + kb);
        const v16bf a0h = rp_load_a(a0hp + kb);
        const v16bf a0l = rp_load_a(a0lp + kb);
        const v16bf a1h = rp_load_a(a1hp + kb);
        const v16bf a1l = rp_load_a(a1lp + kb);

        acc0 = rp_wmma(a0h, bh, acc0);
        acc0 = rp_wmma(a0h, bl, acc0);
        acc0 = rp_wmma(a0l, bh, acc0);
        acc1 = rp_wmma(a1h, bh, acc1);
        acc1 = rp_wmma(a1h, bl, acc1);
        acc1 = rp_wmma(a1l, bh, acc1);
    }

#pragma unroll
    for (int r = 0; r < 8; ++r) {
        const int m = r + 8 * half;
        out[(size_t)m        * O_ + nbase + col] = acc0[r];
        out[(size_t)(16 + m) * O_ + nbase + col] = acc1[r];
    }
}

// ---------------------------------------------------------------------------
extern "C" void kernel_launch(void* const* d_in, const int* in_sizes, int n_in,
                              void* d_out, int out_size, void* d_ws, size_t ws_size,
                              hipStream_t stream) {
    const float* x = (const float*)d_in[0];   // [32, 512, 768]
    const float* p = (const float*)d_in[1];   // [2048, 768]
    float* out = (float*)d_out;               // [32, 2048]

    char* ws = (char*)d_ws;
    float* m_sum = (float*)ws;                         // 32*768 f32   =  98304 B
    bf16*  m_hi  = (bf16*)(ws + 98304);                // 32*768 bf16  =  49152 B
    bf16*  m_lo  = (bf16*)(ws + 147456);               // 32*768 bf16  =  49152 B
    bf16*  p_hi  = (bf16*)(ws + 196608);               // 2048*768 bf16 = 3145728 B
    bf16*  p_lo  = (bf16*)(ws + 3342336);              // 2048*768 bf16 = 3145728 B
                                                       // total ~6.2 MB

    hipMemsetAsync(m_sum, 0, (size_t)B_ * D_ * sizeof(float), stream);

    rp_reduce_x  <<<B_ * (S_ / 64), 256, 0, stream>>>(x, m_sum);
    rp_norm_p    <<<O_ / 8,         256, 0, stream>>>(p, p_hi, p_lo);
    rp_finalize_m<<<(B_ * D_) / 256,256, 0, stream>>>(m_sum, m_hi, m_lo);
    rp_gemm      <<<O_ / 16 / 4,    128, 0, stream>>>(m_hi, m_lo, p_hi, p_lo, out);
}
